// MultiHeadAttention_6992206758195
// MI455X (gfx1250) — matrix operand order repaired and checked by reading, not compiled, not run on hardware
//
#include <hip/hip_runtime.h>

// ---------------------------------------------------------------------------
// MI455X (gfx1250, wave32) multi-head attention, all GEMMs via
// v_wmma_f32_16x16x32_f16. One wave owns one 16x64 output tile.
// ---------------------------------------------------------------------------

typedef _Float16 f16;
typedef _Float16 v16h __attribute__((ext_vector_type(16)));
typedef _Float16 v8h  __attribute__((ext_vector_type(8)));
typedef float    v8f  __attribute__((ext_vector_type(8)));

#define NB 8
#define NS 1024
#define NC 1024
#define NH 16
#define ND 64   // DK == DV

__device__ __forceinline__ v16h cat8(v8h lo, v8h hi) {
  return __builtin_shufflevector(lo, hi, 0,1,2,3,4,5,6,7,8,9,10,11,12,13,14,15);
}

__device__ __forceinline__ v8f wmma_f16(v16h a, v16h b, v8f c) {
  // (neg_a, A, neg_b, B, c_mod, C, reuse_a, reuse_b)
  return __builtin_amdgcn_wmma_f32_16x16x32_f16(false, a, false, b, (short)0, c,
                                                false, false);
}

// ---------------------------------------------------------------------------
// 1) fp32 [batch][R][Cc] -> f16 [batch][Cc][R]   (weight transpose + convert)
// ---------------------------------------------------------------------------
__global__ void k_transpose_cvt(const float* __restrict__ src,
                                f16* __restrict__ dst, int R, int Cc) {
  long bt = blockIdx.y;
  long n  = (long)R * Cc;
  long i  = (long)blockIdx.x * blockDim.x + threadIdx.x;
  if (i < n) {
    long r = i / Cc, c = i % Cc;
    dst[bt * n + c * R + r] = (f16)src[bt * n + r * Cc + c];
  }
}

// ---------------------------------------------------------------------------
// 2) Projection GEMM: X[B*S, C] (fp32) x Wt[h][d][C] (f16) + bias -> f16
//    mode 0: out[b][h][s][d]   (Q, K)
//    mode 1: out[b][h][d][s]   (V transposed, so PV B-frag loads are contiguous)
// ---------------------------------------------------------------------------
__global__ __launch_bounds__(128) void
k_proj(const float* __restrict__ x, const f16* __restrict__ wT,
       const float* __restrict__ bias, f16* __restrict__ out, int mode) {
  const int wave = blockIdx.x * 4 + (threadIdx.x >> 5);
  const int lane = threadIdx.x & 31;
  const int m = lane & 15, half = lane >> 4;
  const int h  = wave & (NH - 1);
  const long row0 = (long)(wave >> 4) * 16;         // row tile in [B*S)

  v8f acc[4];
#pragma unroll
  for (int nt = 0; nt < 4; ++nt) {
    float bv = bias[h * ND + nt * 16 + m];
#pragma unroll
    for (int r = 0; r < 8; ++r) acc[nt][r] = bv;
  }

  const float* xrow  = x + (row0 + m) * (long)NC;
  const f16*   wbase = wT + (long)h * ND * NC;

  for (int k0 = 0; k0 < NC; k0 += 32) {
    // A fragment: lane (m, half): K = k0+16*half+0..15
    union { v16h v; f16 e[16]; } a;
    const float* p0 = xrow + k0 + half * 16;
    const float* p1 = xrow + k0 + half * 16 + 8;
#pragma unroll
    for (int j = 0; j < 8; ++j) { a.e[j] = (f16)p0[j]; a.e[8 + j] = (f16)p1[j]; }
#pragma unroll
    for (int nt = 0; nt < 4; ++nt) {
      // B fragment: lane n = m, K = k0 + half*16 + 0..15 (contiguous 32B)
      const f16* wp = wbase + (long)(nt * 16 + m) * NC + k0 + half * 16;
      acc[nt] = wmma_f16(a.v, *(const v16h*)wp, acc[nt]);
    }
  }

  const long b  = row0 >> 10;       // / NS
  const long s0 = row0 & (NS - 1);
#pragma unroll
  for (int nt = 0; nt < 4; ++nt) {
    int d = nt * 16 + m;
#pragma unroll
    for (int r = 0; r < 8; ++r) {
      long s = s0 + r + 8 * half;
      f16 val = (f16)acc[nt][r];
      if (mode == 0) out[(((b * NH) + h) * NS + s) * ND + d] = val;
      else           out[(((b * NH) + h) * ND + d) * (long)NS + s] = val;
    }
  }
}

// ---------------------------------------------------------------------------
// 3) Causal flash attention. One wave: one (b,h) 16-query block over all keys.
//    Q,K: [B,H,S,64] f16    VT: [B,H,64,S] f16    heads: [B,S,H*64] f16
// ---------------------------------------------------------------------------
__global__ __launch_bounds__(128) void
k_attn(const f16* __restrict__ Q, const f16* __restrict__ K,
       const f16* __restrict__ VT, f16* __restrict__ heads) {
  __shared__ __align__(16) f16 plds[4][16][40];   // P tile per wave, padded rows
  const int wslot = threadIdx.x >> 5;
  const int wave  = blockIdx.x * 4 + wslot;
  const int lane  = threadIdx.x & 31;
  const int m = lane & 15, half = lane >> 4;

  const int qt = wave & (NS / 16 - 1);       // 64 query tiles
  const int bh = wave >> 6;                  // b*NH + h
  const int qbase = qt * 16;

  const f16* Qb = Q  + (long)bh * NS * ND;
  const f16* Kb = K  + (long)bh * NS * ND;
  const f16* Vb = VT + (long)bh * ND * NS;

  // Q A-fragments for K-chunks 0..31 and 32..63
  v16h aq[2];
  {
    const f16* qrow = Qb + (long)(qbase + m) * ND;
#pragma unroll
    for (int c = 0; c < 2; ++c)
      aq[c] = cat8(*(const v8h*)(qrow + c * 32 + half * 16),
                   *(const v8h*)(qrow + c * 32 + half * 16 + 8));
  }

  float mrun[8], lrun[8];
  v8f acc[4];
#pragma unroll
  for (int r = 0; r < 8; ++r) { mrun[r] = -3.0e38f; lrun[r] = 0.0f; }
#pragma unroll
  for (int nt = 0; nt < 4; ++nt)
#pragma unroll
    for (int r = 0; r < 8; ++r) acc[nt][r] = 0.0f;

  const int tend = qbase + 16;               // causal: keys < qbase+16
  for (int t0 = 0; t0 < tend; t0 += 32) {
    if (t0 + 32 < tend) {                    // prefetch next K/V tiles
      __builtin_prefetch(Kb + (long)(t0 + 32 + m) * ND, 0, 0);
      __builtin_prefetch(Vb + (long)m * NS + t0 + 32, 0, 0);
    }
    // ---- scores: two 16-key subtiles, DK=64 as two K=32 WMMAs each ----
    v8f c01[2];
#pragma unroll
    for (int sub = 0; sub < 2; ++sub) {
      v8f c = {0.f, 0.f, 0.f, 0.f, 0.f, 0.f, 0.f, 0.f};
      const f16* krow = Kb + (long)(t0 + sub * 16 + m) * ND;
#pragma unroll
      for (int ch = 0; ch < 2; ++ch)
        c = wmma_f16(aq[ch], *(const v16h*)(krow + ch * 32 + half * 16), c);
      c01[sub] = c;
    }
    // ---- scale + causal mask (boundary blocks only) ----
    const bool boundary = (t0 + 31 > qbase);
#pragma unroll
    for (int sub = 0; sub < 2; ++sub)
#pragma unroll
      for (int r = 0; r < 8; ++r) {
        float sv = c01[sub][r] * 0.125f;     // 1/sqrt(64)
        if (boundary) {
          int key = t0 + sub * 16 + m;
          int qrw = qbase + r + 8 * half;
          if (key > qrw) sv = -3.0e38f;
        }
        c01[sub][r] = sv;
      }
    // ---- online softmax (rows striped over 16-lane halves) ----
#pragma unroll
    for (int r = 0; r < 8; ++r) {
      float tm = fmaxf(c01[0][r], c01[1][r]);
      tm = fmaxf(tm, __shfl_xor(tm, 1));
      tm = fmaxf(tm, __shfl_xor(tm, 2));
      tm = fmaxf(tm, __shfl_xor(tm, 4));
      tm = fmaxf(tm, __shfl_xor(tm, 8));
      float newm  = fmaxf(mrun[r], tm);
      float alpha = __expf(mrun[r] - newm);
      mrun[r] = newm;
      float p0 = __expf(c01[0][r] - newm);
      float p1 = __expf(c01[1][r] - newm);
      c01[0][r] = p0; c01[1][r] = p1;
      float rs = p0 + p1;
      rs += __shfl_xor(rs, 1); rs += __shfl_xor(rs, 2);
      rs += __shfl_xor(rs, 4); rs += __shfl_xor(rs, 8);
      lrun[r] = lrun[r] * alpha + rs;
#pragma unroll
      for (int nt = 0; nt < 4; ++nt) acc[nt][r] *= alpha;
    }
    // ---- transpose P (C-layout -> A-layout) through LDS ----
#pragma unroll
    for (int r = 0; r < 8; ++r) {
      int row = r + 8 * half;
      plds[wslot][row][m]      = (f16)c01[0][r];
      plds[wslot][row][16 + m] = (f16)c01[1][r];
    }
    asm volatile("s_wait_dscnt 0" ::: "memory");   // same-wave LDS RAW
    const f16* pr = &plds[wslot][m][0];
    v16h pa = cat8(*(const v8h*)(pr + 16 * half),
                   *(const v8h*)(pr + 16 * half + 8));
    // ---- O += P x V  (VT rows are contiguous in the key dim) ----
#pragma unroll
    for (int nt = 0; nt < 4; ++nt) {
      const f16* vrow = Vb + (long)(nt * 16 + m) * NS + t0 + half * 16;
      acc[nt] = wmma_f16(pa, *(const v16h*)vrow, acc[nt]);
    }
  }

  // ---- normalize + store concatenated heads [B,S,H*64] ----
  const long b = bh >> 4;
  const int  h = bh & 15;
#pragma unroll
  for (int nt = 0; nt < 4; ++nt) {
    int d = nt * 16 + m;
#pragma unroll
    for (int r = 0; r < 8; ++r) {
      int s = qbase + r + 8 * half;
      float o = acc[nt][r] / lrun[r];
      heads[(b * NS + s) * (long)(NH * ND) + h * ND + d] = (f16)o;
    }
  }
}

// ---------------------------------------------------------------------------
// 4) Output projection: heads[B*S,1024] x WoT[1024][1024] + bo -> fp32 out
// ---------------------------------------------------------------------------
__global__ __launch_bounds__(128) void
k_oproj(const f16* __restrict__ heads, const f16* __restrict__ woT,
        const float* __restrict__ bo, float* __restrict__ out) {
  const int wave = blockIdx.x * 4 + (threadIdx.x >> 5);
  const int lane = threadIdx.x & 31;
  const int m = lane & 15, half = lane >> 4;
  const int  ct   = wave & 15;               // 16 column tiles of 64
  const long row0 = (long)(wave >> 4) * 16;  // 512 row tiles

  v8f acc[4];
#pragma unroll
  for (int nt = 0; nt < 4; ++nt) {
    float bv = bo[ct * 64 + nt * 16 + m];
#pragma unroll
    for (int r = 0; r < 8; ++r) acc[nt][r] = bv;
  }

  const f16* arow = heads + (row0 + m) * (long)(NH * ND);
  for (int k0 = 0; k0 < NH * ND; k0 += 32) {
    v16h a = cat8(*(const v8h*)(arow + k0 + half * 16),
                  *(const v8h*)(arow + k0 + half * 16 + 8));
#pragma unroll
    for (int nt = 0; nt < 4; ++nt) {
      const f16* wp = woT + (long)(ct * 64 + nt * 16 + m) * (NH * ND)
                    + k0 + half * 16;
      acc[nt] = wmma_f16(a, *(const v16h*)wp, acc[nt]);
    }
  }
#pragma unroll
  for (int nt = 0; nt < 4; ++nt) {
    int n = ct * 64 + nt * 16 + m;
#pragma unroll
    for (int r = 0; r < 8; ++r)
      out[(row0 + r + 8 * half) * (long)NC + n] = acc[nt][r];
  }
}

// ---------------------------------------------------------------------------
extern "C" void kernel_launch(void* const* d_in, const int* in_sizes, int n_in,
                              void* d_out, int out_size, void* d_ws,
                              size_t ws_size, hipStream_t stream) {
  const float* q_embs = (const float*)d_in[0];
  const float* k_embs = (const float*)d_in[1];
  const float* v_embs = (const float*)d_in[2];
  const float* Wq = (const float*)d_in[3];
  const float* bq = (const float*)d_in[4];
  const float* Wk = (const float*)d_in[5];
  const float* bk = (const float*)d_in[6];
  const float* Wv = (const float*)d_in[7];
  const float* bv = (const float*)d_in[8];
  const float* Wo = (const float*)d_in[9];
  const float* bo = (const float*)d_in[10];
  float* out = (float*)d_out;

  // f16 workspace layout (elements): 4x 1M weights + 4x 8M activations = 72 MB
  f16* ws  = (f16*)d_ws;
  const size_t WSZ = (size_t)NH * ND * NC;          // 1M
  const size_t ASZ = (size_t)NB * NH * NS * ND;     // 8M
  f16* WqT = ws;
  f16* WkT = WqT + WSZ;
  f16* WvT = WkT + WSZ;
  f16* WoT = WvT + WSZ;
  f16* Qp  = WoT + WSZ;
  f16* Kp  = Qp + ASZ;
  f16* VTp = Kp + ASZ;
  f16* Hp  = VTp + ASZ;

  // 1) weight transpose + fp32->f16
  dim3 tg((NC * ND + 255) / 256, NH);
  k_transpose_cvt<<<tg, 256, 0, stream>>>(Wq, WqT, NC, ND);
  k_transpose_cvt<<<tg, 256, 0, stream>>>(Wk, WkT, NC, ND);
  k_transpose_cvt<<<tg, 256, 0, stream>>>(Wv, WvT, NC, ND);
  k_transpose_cvt<<<dim3((NC * NC + 255) / 256, 1), 256, 0, stream>>>(
      Wo, WoT, NH * ND, NC);

  // 2) projections: 8192 waves, 4 waves/block
  k_proj<<<2048, 128, 0, stream>>>(q_embs, WqT, bq, Qp, 0);
  k_proj<<<2048, 128, 0, stream>>>(k_embs, WkT, bk, Kp, 0);
  k_proj<<<2048, 128, 0, stream>>>(v_embs, WvT, bv, VTp, 1);

  // 3) causal flash attention: 8192 waves
  k_attn<<<2048, 128, 0, stream>>>(Qp, Kp, VTp, Hp);

  // 4) output projection: 8192 waves
  k_oproj<<<2048, 128, 0, stream>>>(Hp, WoT, bo, out);
}
